// TransformerEncoder_39487929319878
// MI455X (gfx1250) — matrix operand order repaired and checked by reading, not compiled, not run on hardware
//
#include <hip/hip_runtime.h>

// ---------------------------------------------------------------------------
// CDNA5 (gfx1250, wave32) transformer encoder forward.
// All matrix math on v_wmma_f32_16x16x32_f16 with f32 accumulation.
// WMMA operands are produced in f16 at creation time, so hot loops are pure
// contiguous b128 fragment loads + WMMA, double-buffered (ping-pong register
// sets, 2x unrolled k-loop -> zero rotation moves).
// ---------------------------------------------------------------------------

typedef _Float16 half_t;
typedef __attribute__((ext_vector_type(16))) _Float16 v16h;
typedef __attribute__((ext_vector_type(2)))  _Float16 h2v;
typedef __attribute__((ext_vector_type(8)))  float    v8f;

constexpr int Lnum = 6, Hn = 8, Dm = 512, Fm = 2048;
constexpr int Ssz = 1024, Bsz = 2, DHd = 64;
constexpr int SB  = Ssz * Bsz;                 // 2048 token rows
constexpr size_t SBD = (size_t)SB * Dm;        // 1,048,576

enum { MODE_F32 = 0, MODE_F16 = 1, MODE_F16_RELU = 2, MODE_VT = 3 };

// ---- WMMA fragment loaders (contiguous k = half*16 + s, same order in A and B) -----------------

// A 16x32 f16: lane row M=lane&15; K = half*16 + v*2 (+0/1)
// -> two contiguous 16B chunks per lane -> 2x global_load_b128.
__device__ __forceinline__ v16h load_frag_A_h(const half_t* __restrict__ p, int half) {
  v16h f;
#pragma unroll
  for (int v = 0; v < 8; ++v) {
    int k = (half << 4) + (v << 1);
    h2v t = *(const h2v*)(p + k);
    f[2 * v]     = t.x;
    f[2 * v + 1] = t.y;
  }
  return f;
}

// B 32x16 f16 from (N,K)-major memory: lane col N fixed, K = half*16 + 0..15
// -> one aligned contiguous 32-byte vector load.
__device__ __forceinline__ v16h load_frag_B(const half_t* __restrict__ rowN, int k0, int half) {
  return *(const v16h*)(rowN + k0 + (half << 4));
}

__device__ __forceinline__ v8f wmma16(v16h a, v16h b, v8f c) {
  return __builtin_amdgcn_wmma_f32_16x16x32_f16(false, a, false, b, (short)0, c,
                                                false, false);
}

// ---- f32 -> f16 elementwise (initial h16 mirror) --------------------------
__global__ __launch_bounds__(256)
void f32_to_f16_kernel(const float* __restrict__ in, half_t* __restrict__ out) {
  const size_t base = (size_t)blockIdx.x * 2048 + threadIdx.x;
#pragma unroll
  for (int i = 0; i < 8; ++i) out[base + i * 256] = (half_t)in[base + i * 256];
}

// ---- weight transpose + f16 convert: W(K,N) f32 -> Wt(N,K) f16 ------------
__global__ __launch_bounds__(256)
void transpose_f16_kernel(const float* __restrict__ W, half_t* __restrict__ Wt,
                          int K, int N) {
  __shared__ half_t t[64][65];
  const int n0 = blockIdx.x * 64, k0 = blockIdx.y * 64;
  const int c = threadIdx.x & 63, r0 = threadIdx.x >> 6;
#pragma unroll
  for (int i = 0; i < 16; ++i) {
    int r = r0 * 16 + i;
    t[c][r] = (half_t)W[(size_t)(k0 + r) * N + n0 + c];
  }
  __syncthreads();
#pragma unroll
  for (int i = 0; i < 16; ++i) {
    int r = r0 * 16 + i;
    Wt[(size_t)(n0 + r) * K + k0 + c] = t[r][c];
  }
}

// ---- GEMM: C[M,N] = A[M,K](f16) @ Bt[N,K](f16)^T + bias -------------------
// 256 threads = 8 waves; wave tile 64M x 32N (8 accs); block tile 128M x 128N.
// Ping-pong double buffering (K % 64 == 0 for all call sites).
__global__ __launch_bounds__(256)
void gemm_tn_kernel(const half_t* __restrict__ A, const half_t* __restrict__ Bt,
                    const float* __restrict__ bias, void* __restrict__ Cout,
                    int M, int N, int K, int mode) {
  const int lane = threadIdx.x & 31, wave = threadIdx.x >> 5;
  const int half = lane >> 4, lm = lane & 15;
  const int m0 = blockIdx.y * 128 + (wave >> 2) * 64;
  const int n0 = blockIdx.x * 128 + (wave & 3) * 32;

  const half_t* aP0 = A + (size_t)(m0 + 0 * 16 + lm) * K;
  const half_t* aP1 = A + (size_t)(m0 + 1 * 16 + lm) * K;
  const half_t* aP2 = A + (size_t)(m0 + 2 * 16 + lm) * K;
  const half_t* aP3 = A + (size_t)(m0 + 3 * 16 + lm) * K;
  const half_t* bR0 = Bt + (size_t)(n0 + lm) * K;
  const half_t* bR1 = Bt + (size_t)(n0 + 16 + lm) * K;

  v8f acc[4][2] = {};
  v16h a0[4], b0[2], a1[4], b1[2];

  auto loadset = [&](v16h (&fa)[4], v16h (&fb)[2], int kk) {
    fb[0] = load_frag_B(bR0, kk, half);
    fb[1] = load_frag_B(bR1, kk, half);
    fa[0] = load_frag_A_h(aP0 + kk, half);
    fa[1] = load_frag_A_h(aP1 + kk, half);
    fa[2] = load_frag_A_h(aP2 + kk, half);
    fa[3] = load_frag_A_h(aP3 + kk, half);
  };
  auto domma = [&](v16h (&fa)[4], v16h (&fb)[2]) {
#pragma unroll
    for (int i = 0; i < 4; ++i) {
      acc[i][0] = wmma16(fa[i], fb[0], acc[i][0]);
      acc[i][1] = wmma16(fa[i], fb[1], acc[i][1]);
    }
  };

  loadset(a0, b0, 0);
  int k0 = 0;
  for (; k0 < K - 64; k0 += 64) {
    loadset(a1, b1, k0 + 32);
    domma(a0, b0);
    loadset(a0, b0, k0 + 64);
    domma(a1, b1);
  }
  loadset(a1, b1, k0 + 32);
  domma(a0, b0);
  domma(a1, b1);

#pragma unroll
  for (int j = 0; j < 2; ++j) {
    const int n = n0 + j * 16 + lm;
    const float bv = bias[n];
#pragma unroll
    for (int i = 0; i < 4; ++i)
#pragma unroll
      for (int r = 0; r < 8; ++r) {
        const int m = m0 + i * 16 + half * 8 + r;
        float xv = acc[i][j][r] + bv;
        if (mode == MODE_F16_RELU) xv = fmaxf(xv, 0.0f);
        if (mode == MODE_F16 || mode == MODE_F16_RELU) {
          ((half_t*)Cout)[(size_t)m * N + n] = (half_t)xv;
        } else if (mode == MODE_VT) {
          // V^T layout: vt[(b*H+hh)][dh][s], m = s*B + b, n = hh*64 + dh
          const int s = m >> 1, b = m & 1, hh = n >> 6, dh = n & 63;
          ((half_t*)Cout)[(((size_t)(b * Hn + hh)) * DHd + dh) * Ssz + s] = (half_t)xv;
        } else {
          ((float*)Cout)[(size_t)m * N + n] = xv;
        }
      }
  }
}

// ---- Attention scores: attn[bh][s][t] = (Q.K)/8, Q/K f16 ------------------
// wave tile 64M x 32N; block tile 128 x 128; grid (S/128, S/128, B*H).
__global__ __launch_bounds__(256)
void attn_scores_kernel(const half_t* __restrict__ Q, const half_t* __restrict__ Kb,
                        float* __restrict__ attn) {
  const int bh = blockIdx.z, b = bh / Hn, hh = bh % Hn, cbase = hh * DHd;
  const int lane = threadIdx.x & 31, wave = threadIdx.x >> 5;
  const int half = lane >> 4, lm = lane & 15;
  const int m0 = blockIdx.y * 128 + (wave >> 2) * 64;
  const int n0 = blockIdx.x * 128 + (wave & 3) * 32;

  const half_t* aP0 = Q + ((size_t)(m0 + 0 * 16 + lm) * Bsz + b) * Dm + cbase;
  const half_t* aP1 = Q + ((size_t)(m0 + 1 * 16 + lm) * Bsz + b) * Dm + cbase;
  const half_t* aP2 = Q + ((size_t)(m0 + 2 * 16 + lm) * Bsz + b) * Dm + cbase;
  const half_t* aP3 = Q + ((size_t)(m0 + 3 * 16 + lm) * Bsz + b) * Dm + cbase;
  const half_t* bR0 = Kb + ((size_t)(n0 + lm) * Bsz + b) * Dm + cbase;
  const half_t* bR1 = Kb + ((size_t)(n0 + 16 + lm) * Bsz + b) * Dm + cbase;

  v8f acc[4][2] = {};
#pragma unroll
  for (int k0 = 0; k0 < DHd; k0 += 32) {
    v16h bf0 = load_frag_B(bR0, k0, half);
    v16h bf1 = load_frag_B(bR1, k0, half);
    v16h a0 = load_frag_A_h(aP0 + k0, half);
    acc[0][0] = wmma16(a0, bf0, acc[0][0]);
    acc[0][1] = wmma16(a0, bf1, acc[0][1]);
    v16h a1 = load_frag_A_h(aP1 + k0, half);
    acc[1][0] = wmma16(a1, bf0, acc[1][0]);
    acc[1][1] = wmma16(a1, bf1, acc[1][1]);
    v16h a2 = load_frag_A_h(aP2 + k0, half);
    acc[2][0] = wmma16(a2, bf0, acc[2][0]);
    acc[2][1] = wmma16(a2, bf1, acc[2][1]);
    v16h a3 = load_frag_A_h(aP3 + k0, half);
    acc[3][0] = wmma16(a3, bf0, acc[3][0]);
    acc[3][1] = wmma16(a3, bf1, acc[3][1]);
  }

  float* out = attn + (size_t)bh * Ssz * Ssz;
#pragma unroll
  for (int j = 0; j < 2; ++j)
#pragma unroll
    for (int i = 0; i < 4; ++i)
#pragma unroll
      for (int r = 0; r < 8; ++r) {
        const int m = m0 + i * 16 + half * 8 + r;
        out[(size_t)m * Ssz + n0 + j * 16 + lm] = acc[i][j][r] * 0.125f;
      }
}

// ---- Row softmax, one wave32 per 1024-wide row ----------------------------
// Writes normalized f32 in place (d_out) and an f16 mirror for attn@V.
__global__ __launch_bounds__(256)
void softmax_kernel(float* __restrict__ attn, half_t* __restrict__ attnh) {
  const int lane = threadIdx.x & 31, wave = threadIdx.x >> 5;
  const int row = blockIdx.x * 8 + wave;
  float* p = attn + (size_t)row * Ssz;
  half_t* ph = attnh + (size_t)row * Ssz;

  float vals[32];
  float mx = -3.402823e38f;
#pragma unroll
  for (int i = 0; i < 32; ++i) {
    vals[i] = p[i * 32 + lane];
    mx = fmaxf(mx, vals[i]);
  }
#pragma unroll
  for (int o = 16; o > 0; o >>= 1) mx = fmaxf(mx, __shfl_xor(mx, o, 32));

  float sum = 0.0f;
#pragma unroll
  for (int i = 0; i < 32; ++i) {
    vals[i] = __expf(vals[i] - mx);
    sum += vals[i];
  }
#pragma unroll
  for (int o = 16; o > 0; o >>= 1) sum += __shfl_xor(sum, o, 32);

  const float inv = 1.0f / sum;
#pragma unroll
  for (int i = 0; i < 32; ++i) {
    const float a = vals[i] * inv;
    p[i * 32 + lane] = a;
    ph[i * 32 + lane] = (half_t)a;
  }
}

// ---- attn(f16) @ V^T(f16) -> O(f16) in (S*B, D) layout --------------------
// wave tile 64M x 16N; block tile 128 x 64; grid (1, S/128, B*H). Ping-pong.
__global__ __launch_bounds__(256)
void attn_apply_kernel(const half_t* __restrict__ attnh, const half_t* __restrict__ vt,
                       half_t* __restrict__ O) {
  const int bh = blockIdx.z, b = bh / Hn, hh = bh % Hn, cbase = hh * DHd;
  const int lane = threadIdx.x & 31, wave = threadIdx.x >> 5;
  const int half = lane >> 4, lm = lane & 15;
  const int m0 = blockIdx.y * 128 + (wave >> 2) * 64;
  const int nIdx = (wave & 3) * 16 + lm;

  const half_t* base = attnh + (size_t)bh * Ssz * Ssz;
  const half_t* aP0 = base + (size_t)(m0 + 0 * 16 + lm) * Ssz;
  const half_t* aP1 = base + (size_t)(m0 + 1 * 16 + lm) * Ssz;
  const half_t* aP2 = base + (size_t)(m0 + 2 * 16 + lm) * Ssz;
  const half_t* aP3 = base + (size_t)(m0 + 3 * 16 + lm) * Ssz;
  const half_t* bR = vt + ((size_t)bh * DHd + nIdx) * Ssz;

  v8f acc[4] = {};
  v16h a0[4], a1[4], b0, b1;

  auto loadset = [&](v16h (&fa)[4], v16h& fb, int kk) {
    fb = load_frag_B(bR, kk, half);
    fa[0] = load_frag_A_h(aP0 + kk, half);
    fa[1] = load_frag_A_h(aP1 + kk, half);
    fa[2] = load_frag_A_h(aP2 + kk, half);
    fa[3] = load_frag_A_h(aP3 + kk, half);
  };
  auto domma = [&](v16h (&fa)[4], v16h& fb) {
#pragma unroll
    for (int i = 0; i < 4; ++i) acc[i] = wmma16(fa[i], fb, acc[i]);
  };

  loadset(a0, b0, 0);
  int k0 = 0;
  for (; k0 < Ssz - 64; k0 += 64) {
    loadset(a1, b1, k0 + 32);
    domma(a0, b0);
    loadset(a0, b0, k0 + 64);
    domma(a1, b1);
  }
  loadset(a1, b1, k0 + 32);
  domma(a0, b0);
  domma(a1, b1);

#pragma unroll
  for (int i = 0; i < 4; ++i)
#pragma unroll
    for (int r = 0; r < 8; ++r) {
      const int m = m0 + i * 16 + half * 8 + r;
      O[((size_t)m * Bsz + b) * Dm + cbase + nIdx] = (half_t)acc[i][r];
    }
}

// ---- residual add + LayerNorm, one wave32 per 512-wide row ----------------
// Writes f32 output and (optionally) an f16 mirror for downstream GEMMs.
__global__ __launch_bounds__(256)
void residual_ln_kernel(const float* __restrict__ hin, const float* __restrict__ add,
                        const float* __restrict__ g, const float* __restrict__ bt,
                        float* __restrict__ out, half_t* __restrict__ out16) {
  const int lane = threadIdx.x & 31, wave = threadIdx.x >> 5;
  const int row = blockIdx.x * 8 + wave;
  const float* hp = hin + (size_t)row * Dm;
  const float* ap = add ? add + (size_t)row * Dm : nullptr;

  float x[16];
  float s = 0.0f;
#pragma unroll
  for (int i = 0; i < 16; ++i) {
    int idx = i * 32 + lane;
    x[i] = hp[idx] + (ap ? ap[idx] : 0.0f);
    s += x[i];
  }
#pragma unroll
  for (int o = 16; o > 0; o >>= 1) s += __shfl_xor(s, o, 32);
  const float mu = s * (1.0f / Dm);

  float vs = 0.0f;
#pragma unroll
  for (int i = 0; i < 16; ++i) {
    float d = x[i] - mu;
    vs += d * d;
  }
#pragma unroll
  for (int o = 16; o > 0; o >>= 1) vs += __shfl_xor(vs, o, 32);
  const float inv = rsqrtf(vs * (1.0f / Dm) + 1e-5f);

#pragma unroll
  for (int i = 0; i < 16; ++i) {
    int idx = i * 32 + lane;
    const float y = (x[i] - mu) * inv * g[idx] + bt[idx];
    out[(size_t)row * Dm + idx] = y;
    if (out16) out16[(size_t)row * Dm + idx] = (half_t)y;
  }
}

// ---------------------------------------------------------------------------
extern "C" void kernel_launch(void* const* d_in, const int* in_sizes, int n_in,
                              void* d_out, int out_size, void* d_ws, size_t ws_size,
                              hipStream_t stream) {
  const float* x    = (const float*)d_in[0];
  const float* Wq   = (const float*)d_in[1];
  const float* bq   = (const float*)d_in[2];
  const float* Wk   = (const float*)d_in[3];
  const float* bk   = (const float*)d_in[4];
  const float* Wv   = (const float*)d_in[5];
  const float* bv   = (const float*)d_in[6];
  const float* Wo   = (const float*)d_in[7];
  const float* bo   = (const float*)d_in[8];
  const float* ln1g = (const float*)d_in[9];
  const float* ln1b = (const float*)d_in[10];
  const float* W1   = (const float*)d_in[11];
  const float* b1   = (const float*)d_in[12];
  const float* W2   = (const float*)d_in[13];
  const float* b2   = (const float*)d_in[14];
  const float* ln2g = (const float*)d_in[15];
  const float* ln2b = (const float*)d_in[16];
  const float* lnfg = (const float*)d_in[17];
  const float* lnfb = (const float*)d_in[18];

  // f32 workspace region
  float* h   = (float*)d_ws;                 // (S*B, D) residual stream
  float* tmp = h + SBD;                      // (S*B, D) proj / ff2 out
  // f16 workspace region
  half_t* h16  = (half_t*)(tmp + SBD);                // f16 mirror of h
  half_t* qh   = h16 + SBD;                           // (S*B, D)
  half_t* kh   = qh + SBD;                            // (S*B, D)
  half_t* vt   = kh + SBD;                            // (B*H, DH, S)
  half_t* obh  = vt + SBD;                            // (S*B, D) attn out
  half_t* ff1h = obh + SBD;                           // (S*B, F)
  half_t* wt   = ff1h + (size_t)SB * Fm;              // transposed weight buf
  half_t* attnh = wt + (size_t)Fm * Dm;               // (B*H, S, S) f16 attn

  float* outx  = (float*)d_out;
  float* attn0 = outx + SBD;                 // (L, B*H, S, S)

  (void)hipMemcpyAsync(h, x, SBD * sizeof(float), hipMemcpyDeviceToDevice, stream);

  const dim3 blk(256);
  const dim3 gCvt(SBD / 2048);
  const dim3 gTD(Dm / 64, Dm / 64);          // transpose 512x512
  const dim3 gT1(Fm / 64, Dm / 64);          // transpose W1 (K=512,N=2048)
  const dim3 gT2(Dm / 64, Fm / 64);          // transpose W2 (K=2048,N=512)
  const dim3 gD(Dm / 128, SB / 128);         // GEMM out (2048 x 512)
  const dim3 gF(Fm / 128, SB / 128);         // GEMM out (2048 x 2048)
  const dim3 gScore(Ssz / 128, Ssz / 128, Bsz * Hn);
  const dim3 gApply(1, Ssz / 128, Bsz * Hn);
  const dim3 gSoft((Bsz * Hn * Ssz) / 8);
  const dim3 gLN(SB / 8);

  f32_to_f16_kernel<<<gCvt, blk, 0, stream>>>(h, h16);

  for (int l = 0; l < Lnum; ++l) {
    const float* wq = Wq + (size_t)l * Dm * Dm;  const float* bql = bq + (size_t)l * Dm;
    const float* wk = Wk + (size_t)l * Dm * Dm;  const float* bkl = bk + (size_t)l * Dm;
    const float* wv = Wv + (size_t)l * Dm * Dm;  const float* bvl = bv + (size_t)l * Dm;
    const float* wo = Wo + (size_t)l * Dm * Dm;  const float* bol = bo + (size_t)l * Dm;
    const float* w1 = W1 + (size_t)l * Dm * Fm;  const float* b1l = b1 + (size_t)l * Fm;
    const float* w2 = W2 + (size_t)l * Fm * Dm;  const float* b2l = b2 + (size_t)l * Dm;
    float* attnL = attn0 + (size_t)l * Bsz * Hn * Ssz * Ssz;

    transpose_f16_kernel<<<gTD, blk, 0, stream>>>(wq, wt, Dm, Dm);
    gemm_tn_kernel<<<gD, blk, 0, stream>>>(h16, wt, bql, qh, SB, Dm, Dm, MODE_F16);
    transpose_f16_kernel<<<gTD, blk, 0, stream>>>(wk, wt, Dm, Dm);
    gemm_tn_kernel<<<gD, blk, 0, stream>>>(h16, wt, bkl, kh, SB, Dm, Dm, MODE_F16);
    transpose_f16_kernel<<<gTD, blk, 0, stream>>>(wv, wt, Dm, Dm);
    gemm_tn_kernel<<<gD, blk, 0, stream>>>(h16, wt, bvl, vt, SB, Dm, Dm, MODE_VT);

    attn_scores_kernel<<<gScore, blk, 0, stream>>>(qh, kh, attnL);
    softmax_kernel<<<gSoft, blk, 0, stream>>>(attnL, attnh);
    attn_apply_kernel<<<gApply, blk, 0, stream>>>(attnh, vt, obh);

    transpose_f16_kernel<<<gTD, blk, 0, stream>>>(wo, wt, Dm, Dm);
    gemm_tn_kernel<<<gD, blk, 0, stream>>>(obh, wt, bol, tmp, SB, Dm, Dm, MODE_F32);
    residual_ln_kernel<<<gLN, blk, 0, stream>>>(h, tmp, ln1g + (size_t)l * Dm,
                                                ln1b + (size_t)l * Dm, h, h16);

    transpose_f16_kernel<<<gT1, blk, 0, stream>>>(w1, wt, Dm, Fm);
    gemm_tn_kernel<<<gF, blk, 0, stream>>>(h16, wt, b1l, ff1h, SB, Fm, Dm, MODE_F16_RELU);
    transpose_f16_kernel<<<gT2, blk, 0, stream>>>(w2, wt, Fm, Dm);
    gemm_tn_kernel<<<gD, blk, 0, stream>>>(ff1h, wt, b2l, tmp, SB, Dm, Fm, MODE_F32);
    residual_ln_kernel<<<gLN, blk, 0, stream>>>(h, tmp, ln2g + (size_t)l * Dm,
                                                ln2b + (size_t)l * Dm, h, h16);
  }

  residual_ln_kernel<<<gLN, blk, 0, stream>>>(h, nullptr, lnfg, lnfb, outx, nullptr);

  (void)in_sizes; (void)n_in; (void)out_size; (void)ws_size;
}
